// MultiheadSelfAttention_15496242004656
// MI455X (gfx1250) — matrix operand order repaired and checked by reading, not compiled, not run on hardware
//
#include <hip/hip_runtime.h>
#include <hip/hip_bf16.h>
#include <cstdint>

// ---------------------------------------------------------------------------
// MI455X (gfx1250) multi-head self-attention with RoPE.
// Heavy math: v_wmma_f32_16x16x32_bf16.  Tile staging: Tensor Data Mover.
// B=4, T=2048, D=2048, H=16, dk=128.
// ---------------------------------------------------------------------------

typedef __attribute__((ext_vector_type(16))) __bf16    v16bf;
typedef __attribute__((ext_vector_type(8)))  float     v8f;
typedef __attribute__((ext_vector_type(4)))  uint32_t  v4u;
typedef __attribute__((ext_vector_type(8)))  int       v8i;
typedef __attribute__((ext_vector_type(4)))  int       v4i;

#define DMODEL  2048
#define NHEADS  16
#define DK      128
#define SEQ     2048
#define BATCH   4
#define BT      (BATCH * SEQ)      // 8192
#define KDIM    2048
#define NDIM    2048
#define SOFTMAX_SCALE 0.08838834764831845f   // 1/sqrt(128)

__device__ __forceinline__ uint16_t f2bf(float f) {
  union { float f; uint32_t u; } c; c.f = f;
  uint32_t u = c.u;
  uint32_t r = u + 0x7FFFu + ((u >> 16) & 1u);   // round-to-nearest-even
  return (uint16_t)(r >> 16);
}

union Frag { uint4 u[2]; v16bf v; };

// ---------------------------------------------------------------------------
// Tensor Data Mover: 2-D tile load, dword (4B) granularity.
// D# layout per CDNA5 ISA 8.3/8.4:
//  g0: [1:0]=count=1 | [63:32]=lds_addr | [120:64]=global_addr | [127:126]=type=2
//  g1: [17:16]=data_size=2(4B) | [79:48]=tensor_dim0 | [111:80]=tensor_dim1 |
//      [127:112]=tile_dim0 | [143:128]=tile_dim1 | [207:160]=tensor_dim0_stride
// All sizes/strides in dwords.  Issued per-wave (uniform args), TENSORcnt.
// This toolchain exposes the 6-operand builtin: groups 2/3 + extra group are
// zero (2-D tensor: higher dims unused).
// ---------------------------------------------------------------------------
__device__ __forceinline__ void tdm_load_2d(uint32_t lds_addr, const void* gptr,
                                            uint32_t tile_dw, uint32_t tile_rows,
                                            uint32_t tensor_dw, uint32_t tensor_rows,
                                            uint32_t stride_dw) {
  uint64_t ga = (uint64_t)(uintptr_t)gptr;
  v4u g0;
  g0.x = 1u;                                             // count=1
  g0.y = lds_addr;                                       // LDS byte address
  g0.z = (uint32_t)ga;                                   // global_addr[31:0]
  g0.w = (uint32_t)((ga >> 32) & 0x01FFFFFFu) | (2u << 30); // ga[56:32], type=2
  v8i g1;
  g1[0] = (int)(2u << 16);                               // data_size=4B, mask=0
  g1[1] = (int)((tensor_dw & 0xFFFFu) << 16);            // tensor_dim0[15:0]
  g1[2] = (int)((tensor_dw >> 16) | ((tensor_rows & 0xFFFFu) << 16));
  g1[3] = (int)((tensor_rows >> 16) | (tile_dw << 16));  // tile_dim0
  g1[4] = (int)(tile_rows & 0xFFFFu);                    // tile_dim1, tile_dim2=0
  g1[5] = (int)stride_dw;                                // tensor_dim0_stride[31:0]
  g1[6] = 0;
  g1[7] = 0;
  v4i z4 = (v4i)0;
  v8i z8 = (v8i)0;
  __builtin_amdgcn_tensor_load_to_lds(g0, g1, z4, z4, z8, 0);
}

// ---------------------------------------------------------------------------
// Kernel 1: fp32 -> bf16 elementwise convert (4 elems / thread)
// ---------------------------------------------------------------------------
__global__ __launch_bounds__(256)
void cvt_bf16_vec(const float* __restrict__ src, uint16_t* __restrict__ dst, long n4) {
  long i = (long)blockIdx.x * blockDim.x + threadIdx.x;
  if (i < n4) {
    float4 f = ((const float4*)src)[i];
    ushort4 o;
    o.x = f2bf(f.x); o.y = f2bf(f.y); o.z = f2bf(f.z); o.w = f2bf(f.w);
    ((ushort4*)dst)[i] = o;
  }
}

// ---------------------------------------------------------------------------
// Kernel 2/5: C(MxN,f32) = A(MxK,bf16) * W(NxK,bf16)^T      (K=N=2048)
// 256 thr = 8 waves; wave w: rows [bx*128+16w,+16), cols [by*128,+128).
// W tile (128n x 32k) double-buffered in LDS via TDM (wave 0 issues DMA,
// s_wait_tensorcnt(1) => DMA of tile i+1 overlaps WMMA on tile i).
// ---------------------------------------------------------------------------
__global__ __launch_bounds__(256)
void gemm_bf16_nt(const uint16_t* __restrict__ A,
                  const uint16_t* __restrict__ W,
                  float* __restrict__ Out) {
  __shared__ uint16_t ldsW[2][128 * 32];     // 2 x 8 KB
  const int tid   = threadIdx.x;
  const int wave  = tid >> 5;
  const int lane  = tid & 31;
  const int l16   = lane & 15;
  const int khalf = lane >> 4;
  const long mBase = (long)blockIdx.x * 128 + wave * 16;
  const long nBase = (long)blockIdx.y * 128;

  const uint32_t lds0 = (uint32_t)(uintptr_t)&ldsW[0][0];
  const uint32_t lds1 = (uint32_t)(uintptr_t)&ldsW[1][0];
  const uint16_t* wbase = W + nBase * (long)KDIM;

  // prologue: DMA tile 0 into buffer 0
  if (wave == 0)
    tdm_load_2d(lds0, wbase, /*tile_dw*/16, /*tile_rows*/128,
                /*tensor_dw*/KDIM / 2, /*tensor_rows*/NDIM, /*stride_dw*/KDIM / 2);

  v8f acc[8] = {};
  const uint16_t* arow = A + (mBase + l16) * (long)KDIM;

  for (int i = 0; i < 64; ++i) {
    const int k0 = i * 32;
    if (wave == 0) {
      // issue tile i+1 into the other buffer (last iter: redundant re-load)
      const int knext = (i < 63) ? k0 + 32 : k0;
      tdm_load_2d((i & 1) ? lds0 : lds1, wbase + knext, 16, 128,
                  KDIM / 2, NDIM, KDIM / 2);
      __builtin_amdgcn_s_wait_tensorcnt(1);   // tile i complete (in-order)
    }
    __syncthreads();

    const uint16_t* lw = &ldsW[i & 1][0];

    Frag af;
    af.u[0] = *(const uint4*)(arow + k0 + khalf * 8);
    af.u[1] = *(const uint4*)(arow + k0 + 16 + khalf * 8);

    Frag bfr[8];
#pragma unroll
    for (int nt = 0; nt < 8; ++nt) {
      const uint16_t* bp = lw + (nt * 16 + l16) * 32 + khalf * 8;
      bfr[nt].u[0] = *(const uint4*)(bp);
      bfr[nt].u[1] = *(const uint4*)(bp + 16);
    }
#pragma unroll
    for (int nt = 0; nt < 8; ++nt)
      acc[nt] = __builtin_amdgcn_wmma_f32_16x16x32_bf16(
          false, af.v, false, bfr[nt].v, (short)0, acc[nt], false, false);

    __syncthreads();   // all waves done with buffer i before its DMA overwrite
  }

#pragma unroll
  for (int nt = 0; nt < 8; ++nt) {
#pragma unroll
    for (int g = 0; g < 8; ++g) {
      long m = mBase + khalf * 8 + g;
      long n = nBase + nt * 16 + l16;
      Out[m * (long)NDIM + n] = acc[nt][g];
    }
  }
}

// ---------------------------------------------------------------------------
// Kernel 3a: RoPE on Q,K (fp32 [B,T,D]) -> bf16 [B,H,T,dk]
// ---------------------------------------------------------------------------
__global__ __launch_bounds__(256)
void rope_cvt_qk(const float* __restrict__ Qf, const float* __restrict__ Kf,
                 const int* __restrict__ tpos,
                 uint16_t* __restrict__ Qb, uint16_t* __restrict__ Kb) {
  long i = (long)blockIdx.x * blockDim.x + threadIdx.x;
  int  j = (int)(i & 63);
  int  h = (int)((i >> 6) & 15);
  long t = (i >> 10) & 2047;
  long b = i >> 21;

  long src = ((b * SEQ + t) * DMODEL) + h * DK + 2 * j;
  float2 q = *(const float2*)(Qf + src);
  float2 k = *(const float2*)(Kf + src);

  float pos  = (float)tpos[t];
  float freq = __powf(10000.0f, -2.0f * (float)j * (1.0f / (float)DK));
  float s, c;
  __sincosf(pos * freq, &s, &c);

  float qe = q.x * c - q.y * s, qo = q.x * s + q.y * c;
  float ke = k.x * c - k.y * s, ko = k.x * s + k.y * c;

  long dst = (((b * NHEADS + h) * SEQ + t) * DK) + 2 * j;
  *(uint32_t*)(Qb + dst) = (uint32_t)f2bf(qe) | ((uint32_t)f2bf(qo) << 16);
  *(uint32_t*)(Kb + dst) = (uint32_t)f2bf(ke) | ((uint32_t)f2bf(ko) << 16);
}

// ---------------------------------------------------------------------------
// Kernel 3b: V fp32 [B,T,D] -> bf16 Vt [B,H,dk,T]  (LDS-tiled transpose)
// ---------------------------------------------------------------------------
__global__ __launch_bounds__(256)
void v_transpose(const float* __restrict__ Vf, uint16_t* __restrict__ Vt) {
  __shared__ float tile[32][33];
  const int bh = blockIdx.z, b = bh >> 4, h = bh & 15;
  const int t0 = blockIdx.x * 32, f0 = blockIdx.y * 32;
  const int tx = threadIdx.x, ty = threadIdx.y;

  for (int r = ty; r < 32; r += 8)
    tile[r][tx] = Vf[((long)(b * SEQ + t0 + r)) * DMODEL + h * DK + f0 + tx];
  __syncthreads();
  for (int r = ty; r < 32; r += 8)
    Vt[(((long)bh * DK) + f0 + r) * SEQ + t0 + tx] = f2bf(tile[tx][r]);
}

// ---------------------------------------------------------------------------
// Kernel 4: flash attention (full softmax, no mask)
// grid (T/128, B*H), 256 thr (8 waves); wave owns 16 q-rows.
// K and V 128x128 bf16 tiles DMA'd by TDM each s-block.  P (bf16, transposed
// through LDS into A-fragment layout) reuses the K tile region: K is dead
// after the S-GEMM and a block barrier separates the phases.
// ---------------------------------------------------------------------------
__global__ __launch_bounds__(256)
void flash_attn(const uint16_t* __restrict__ Qb, const uint16_t* __restrict__ Kb,
                const uint16_t* __restrict__ Vt, uint16_t* __restrict__ Ob) {
  __shared__ uint16_t ldsKP[128 * 128];   // 32 KB : K tile, then P tiles
  __shared__ uint16_t ldsV [128 * 128];   // 32 KB : V tile
  const int tid = threadIdx.x, wave = tid >> 5, lane = tid & 31;
  const int l16 = lane & 15, khalf = lane >> 4;
  const int qt = blockIdx.x, bh = blockIdx.y;
  const int b = bh >> 4, h = bh & 15;
  const long qRow = (long)qt * 128 + wave * 16;

  const uint16_t* Qp = Qb + (long)bh * SEQ * DK;
  const uint16_t* Kp = Kb + (long)bh * SEQ * DK;
  const uint16_t* Vp = Vt + (long)bh * DK * SEQ;

  const uint32_t ldsKaddr = (uint32_t)(uintptr_t)&ldsKP[0];
  const uint32_t ldsVaddr = (uint32_t)(uintptr_t)&ldsV[0];

  // Q A-fragments for full dk (4 k-steps), kept in registers
  Frag aq[4];
#pragma unroll
  for (int ks = 0; ks < 4; ++ks) {
    const uint16_t* qp = Qp + (qRow + l16) * DK + ks * 32 + khalf * 8;
    aq[ks].u[0] = *(const uint4*)(qp);
    aq[ks].u[1] = *(const uint4*)(qp + 16);
  }

  v8f oacc[8] = {};
  float rowM[8], rowL[8];
#pragma unroll
  for (int g = 0; g < 8; ++g) { rowM[g] = -3.0e38f; rowL[g] = 0.0f; }

  uint16_t* pw = ldsKP + wave * 16 * 128;   // per-wave 16x128 P tile

  for (int sb = 0; sb < SEQ; sb += 128) {
    // ---- TDM: stage K rows [sb,sb+128) x dk  and  V 128 feats x s-window ----
    if (wave == 0) {
      tdm_load_2d(ldsKaddr, Kp + (long)sb * DK, /*tile_dw*/DK / 2, /*rows*/128,
                  /*tensor_dw*/DK / 2, /*tensor_rows*/SEQ, /*stride_dw*/DK / 2);
      tdm_load_2d(ldsVaddr, Vp + sb, /*tile_dw*/64, /*rows*/DK,
                  /*tensor_dw*/SEQ / 2, /*tensor_rows*/DK, /*stride_dw*/SEQ / 2);
      __builtin_amdgcn_s_wait_tensorcnt(0);
    }
    __syncthreads();

    // ---- S = Q K^T (16 x 128 per wave) ----
    v8f sacc[8] = {};
#pragma unroll
    for (int ks = 0; ks < 4; ++ks) {
#pragma unroll
      for (int h4 = 0; h4 < 2; ++h4) {
        Frag bk[4];
#pragma unroll
        for (int q = 0; q < 4; ++q) {
          const int nt = h4 * 4 + q;
          const uint16_t* kp = ldsKP + (nt * 16 + l16) * DK + ks * 32 + khalf * 8;
          bk[q].u[0] = *(const uint4*)(kp);
          bk[q].u[1] = *(const uint4*)(kp + 16);
        }
#pragma unroll
        for (int q = 0; q < 4; ++q)
          sacc[h4 * 4 + q] = __builtin_amdgcn_wmma_f32_16x16x32_bf16(
              false, aq[ks].v, false, bk[q].v, (short)0, sacc[h4 * 4 + q], false, false);
      }
    }
    __syncthreads();   // K tile fully consumed; region becomes P storage

    // ---- online softmax (row = khalf*8+g; cols across lanes & nt tiles) ----
#pragma unroll
    for (int g = 0; g < 8; ++g) {
      float m = -3.0e38f;
#pragma unroll
      for (int nt = 0; nt < 8; ++nt) {
        sacc[nt][g] *= SOFTMAX_SCALE;
        m = fmaxf(m, sacc[nt][g]);
      }
      m = fmaxf(m, __shfl_xor(m, 1, 32));
      m = fmaxf(m, __shfl_xor(m, 2, 32));
      m = fmaxf(m, __shfl_xor(m, 4, 32));
      m = fmaxf(m, __shfl_xor(m, 8, 32));
      float newM = fmaxf(rowM[g], m);
      float fac  = __expf(rowM[g] - newM);
      float s = 0.0f;
#pragma unroll
      for (int nt = 0; nt < 8; ++nt) {
        float p = __expf(sacc[nt][g] - newM);
        sacc[nt][g] = p;
        s += p;
      }
      s += __shfl_xor(s, 1, 32);
      s += __shfl_xor(s, 2, 32);
      s += __shfl_xor(s, 4, 32);
      s += __shfl_xor(s, 8, 32);
      rowL[g] = rowL[g] * fac + s;
      rowM[g] = newM;
#pragma unroll
      for (int nt = 0; nt < 8; ++nt) oacc[nt][g] *= fac;
    }

    // ---- transpose P (bf16) through LDS into A-fragment layout ----
#pragma unroll
    for (int g = 0; g < 8; ++g) {
      int m = khalf * 8 + g;
#pragma unroll
      for (int nt = 0; nt < 8; ++nt)
        pw[m * 128 + nt * 16 + l16] = f2bf(sacc[nt][g]);
    }
    // wave-private P region; same-wave DS ordering makes stores visible

    // ---- O += P V ----
#pragma unroll
    for (int ks = 0; ks < 4; ++ks) {
      Frag ap;
      const uint16_t* pp = pw + l16 * 128 + ks * 32 + khalf * 8;
      ap.u[0] = *(const uint4*)(pp);
      ap.u[1] = *(const uint4*)(pp + 16);
#pragma unroll
      for (int h4 = 0; h4 < 2; ++h4) {
        Frag bv[4];
#pragma unroll
        for (int q = 0; q < 4; ++q) {
          const int nt = h4 * 4 + q;
          const uint16_t* vp = ldsV + (nt * 16 + l16) * 128 + ks * 32 + khalf * 8;
          bv[q].u[0] = *(const uint4*)(vp);
          bv[q].u[1] = *(const uint4*)(vp + 16);
        }
#pragma unroll
        for (int q = 0; q < 4; ++q)
          oacc[h4 * 4 + q] = __builtin_amdgcn_wmma_f32_16x16x32_bf16(
              false, ap.v, false, bv[q].v, (short)0, oacc[h4 * 4 + q], false, false);
      }
    }
    __syncthreads();   // P and V consumed before next s-block's DMA overwrite
  }

  // ---- finalize: O /= l, write bf16 [B,T,D] ----
#pragma unroll
  for (int g = 0; g < 8; ++g) {
    float inv = 1.0f / rowL[g];
    long t = qRow + khalf * 8 + g;
#pragma unroll
    for (int nt = 0; nt < 8; ++nt) {
      long idx = ((long)b * SEQ + t) * DMODEL + h * DK + nt * 16 + l16;
      Ob[idx] = f2bf(oacc[nt][g] * inv);
    }
  }
}

// ---------------------------------------------------------------------------
// launch  (workspace 352 MB; O bf16 aliases dead Qf region)
// ---------------------------------------------------------------------------
extern "C" void kernel_launch(void* const* d_in, const int* in_sizes, int n_in,
                              void* d_out, int out_size, void* d_ws, size_t ws_size,
                              hipStream_t stream) {
  const float* x    = (const float*)d_in[0];
  const int*   tpos = (const int*)d_in[1];
  const float* Wq   = (const float*)d_in[2];
  const float* Wk   = (const float*)d_in[3];
  const float* Wv   = (const float*)d_in[4];
  const float* Wo   = (const float*)d_in[5];

  char* ws = (char*)d_ws;
  const long MB = 1024L * 1024L;
  uint16_t* Xb  = (uint16_t*)(ws + 0);
  uint16_t* Wqb = (uint16_t*)(ws + 32 * MB);
  uint16_t* Wkb = (uint16_t*)(ws + 40 * MB);
  uint16_t* Wvb = (uint16_t*)(ws + 48 * MB);
  uint16_t* Wob = (uint16_t*)(ws + 56 * MB);
  float*    Qf  = (float*)   (ws + 64 * MB);
  float*    Kf  = (float*)   (ws + 128 * MB);
  float*    Vf  = (float*)   (ws + 192 * MB);
  uint16_t* Qb2 = (uint16_t*)(ws + 256 * MB);
  uint16_t* Kb2 = (uint16_t*)(ws + 288 * MB);
  uint16_t* Vtb = (uint16_t*)(ws + 320 * MB);
  uint16_t* Ob  = (uint16_t*)(ws + 64 * MB);   // alias over dead Qf

  {
    long n4x = (long)BT * DMODEL / 4;
    cvt_bf16_vec<<<dim3((unsigned)(n4x / 256)), dim3(256), 0, stream>>>(x, Xb, n4x);
    long n4w = (long)DMODEL * DMODEL / 4;
    cvt_bf16_vec<<<dim3((unsigned)(n4w / 256)), dim3(256), 0, stream>>>(Wq, Wqb, n4w);
    cvt_bf16_vec<<<dim3((unsigned)(n4w / 256)), dim3(256), 0, stream>>>(Wk, Wkb, n4w);
    cvt_bf16_vec<<<dim3((unsigned)(n4w / 256)), dim3(256), 0, stream>>>(Wv, Wvb, n4w);
    cvt_bf16_vec<<<dim3((unsigned)(n4w / 256)), dim3(256), 0, stream>>>(Wo, Wob, n4w);
  }

  dim3 ggrid(BT / 128, NDIM / 128);   // (64,16)
  gemm_bf16_nt<<<ggrid, dim3(256), 0, stream>>>(Xb, Wqb, Qf);
  gemm_bf16_nt<<<ggrid, dim3(256), 0, stream>>>(Xb, Wkb, Kf);
  gemm_bf16_nt<<<ggrid, dim3(256), 0, stream>>>(Xb, Wvb, Vf);

  {
    long nPair = (long)BATCH * SEQ * NHEADS * (DK / 2);
    rope_cvt_qk<<<dim3((unsigned)(nPair / 256)), dim3(256), 0, stream>>>(
        Qf, Kf, tpos, Qb2, Kb2);
    v_transpose<<<dim3(SEQ / 32, DK / 32, BATCH * NHEADS), dim3(32, 8), 0, stream>>>(
        Vf, Vtb);
  }

  flash_attn<<<dim3(SEQ / 128, BATCH * NHEADS), dim3(256), 0, stream>>>(
      Qb2, Kb2, Vtb, Ob);

  gemm_bf16_nt<<<ggrid, dim3(256), 0, stream>>>(Ob, Wob, (float*)d_out);
}
